// GapAwareParticleGNN_25812753449150
// MI455X (gfx1250) — matrix operand order repaired and checked by reading, not compiled, not run on hardware
//
#include <hip/hip_runtime.h>
#include <hip/hip_bf16.h>
#include <math.h>

// ---------------------------------------------------------------------------
// GapAwareParticleGNN forward for MI455X (gfx1250, wave32, WMMA).
// Dense projections -> v_wmma_f32_16x16x32_f16 (f32 accumulate).
//   - LDB templated so B loads use immediate offsets
//   - GUARD templated: column guards (clamp+select) only for the Ncol=5
//     classifier GEMM; all hot GEMMs (Ncol 256/512) are guard-free
//   - 2 N-tiles per wave sharing one A fetch
// Edge softmax/scatter -> float atomics (memory bound, irregular).
// ---------------------------------------------------------------------------

#define HIDN 256
#define NH   8
#define HD   32
#define NCLSS 5

typedef __attribute__((ext_vector_type(16))) _Float16 v16h;
typedef __attribute__((ext_vector_type(8)))  float    v8f;

// ------------------------------ helpers ------------------------------------

__device__ inline void atomicMaxF(float* addr, float val) {
    unsigned int* ua = reinterpret_cast<unsigned int*>(addr);
    unsigned int cur = __float_as_uint(*addr);
    while (__uint_as_float(cur) < val) {
        unsigned int prev = atomicCAS(ua, cur, __float_as_uint(val));
        if (prev == cur) break;
        cur = prev;
    }
}

__global__ void k_fill(float* p, float v, int n) {
    int i = blockIdx.x * blockDim.x + threadIdx.x;
    if (i < n) p[i] = v;
}

// C[r, c] = bias[c] for an HIDN-wide block embedded in a row of width ld
__global__ void k_bias_rows(float* C, const float* __restrict__ bias,
                            int rows, int w, int ld) {
    int i = blockIdx.x * blockDim.x + threadIdx.x;
    if (i >= rows * w) return;
    int r = i / w, c = i % w;
    C[(size_t)r * ld + c] = bias[c];
}

// ------------------------------ WMMA GEMM ----------------------------------
// C[M,Ncol] = A[M,K] @ B[K,Ncol] (+bias) (+add) (+relu), f32 in/out, f16 MACs.
// M must be a multiple of 16 (true for all call sites).
// Each wave owns two adjacent 16x16 output tiles (shared A fetch); 4 waves
// per block -> 128 output columns per block.
// GUARD=false requires Ncol to be a multiple of 32 (no column guards emitted).
template <int LDB, bool GUARD>
__global__ __launch_bounds__(128)
void k_gemm_wmma(const float* __restrict__ A, int lda,
                 const float* __restrict__ B,
                 const float* __restrict__ bias,
                 const float* __restrict__ add, int ldadd,
                 float* __restrict__ C, int ldc,
                 int M, int K, int Ncol, int relu) {
    const int lane = threadIdx.x & 31;
    const int wave = threadIdx.x >> 5;
    const int tm   = blockIdx.x;
    const int tn0  = (blockIdx.y * 4 + wave) * 2;     // first of two col tiles
    if (tn0 * 16 >= Ncol) return;                     // wave-uniform exit

    const int m    = lane & 15;
    const int half = lane >> 4;
    const int row  = tm * 16 + m;
    const float* Arow = A + (size_t)row * lda;

    const int col0 = tn0 * 16 + (lane & 15);
    const int col1 = col0 + 16;
    const bool v0 = GUARD ? (col0 < Ncol) : true;
    const bool v1 = GUARD ? (col1 < Ncol) : true;
    const float* B0 = B + (v0 ? col0 : 0);            // clamped, always in-bounds
    const float* B1 = B + (v1 ? col1 : 0);

    v8f acc0 = {}, acc1 = {};
    const int kfull = K & ~31;

    for (int k0 = 0; k0 < kfull; k0 += 32) {
        // A: lane covers (m, k0+half*16 .. +15)
        const float4* p0 = (const float4*)(Arow + k0 + half * 16);
        const float4* p1 = p0 + 2;
        float4 x0 = p0[0], x1 = p0[1], x2 = p1[0], x3 = p1[1];
        v16h a;
        a[0]  = (_Float16)x0.x; a[1]  = (_Float16)x0.y;
        a[2]  = (_Float16)x0.z; a[3]  = (_Float16)x0.w;
        a[4]  = (_Float16)x1.x; a[5]  = (_Float16)x1.y;
        a[6]  = (_Float16)x1.z; a[7]  = (_Float16)x1.w;
        a[8]  = (_Float16)x2.x; a[9]  = (_Float16)x2.y;
        a[10] = (_Float16)x2.z; a[11] = (_Float16)x2.w;
        a[12] = (_Float16)x3.x; a[13] = (_Float16)x3.y;
        a[14] = (_Float16)x3.z; a[15] = (_Float16)x3.w;

        // B: lane covers (k0+half*16+e, col); LDB constant -> imm offsets
        const float* Bk0 = B0 + (size_t)(k0 + half * 16) * LDB;
        v16h b;
#pragma unroll
        for (int e = 0; e < 16; ++e) {
            float t = Bk0[e * LDB];                   // unconditional load
            if (GUARD) b[e] = v0 ? (_Float16)t : (_Float16)0.f;
            else       b[e] = (_Float16)t;
        }
        acc0 = __builtin_amdgcn_wmma_f32_16x16x32_f16(
            false, a, false, b, (short)0, acc0, false, false);

        const float* Bk1 = B1 + (size_t)(k0 + half * 16) * LDB;
#pragma unroll
        for (int e = 0; e < 16; ++e) {
            float t = Bk1[e * LDB];
            if (GUARD) b[e] = v1 ? (_Float16)t : (_Float16)0.f;
            else       b[e] = (_Float16)t;
        }
        acc1 = __builtin_amdgcn_wmma_f32_16x16x32_f16(
            false, a, false, b, (short)0, acc1, false, false);
    }

    if (kfull < K) {                                  // K tail (input proj K=16)
        v16h a, b;
#pragma unroll
        for (int e = 0; e < 16; ++e) {
            int ka = kfull + half * 16 + e;
            float t = Arow[(ka < K) ? ka : 0];        // clamped load
            a[e] = (ka < K) ? (_Float16)t : (_Float16)0.f;
        }
#pragma unroll
        for (int e = 0; e < 16; ++e) {
            int kb = kfull + half * 16 + e;
            float t = B0[(size_t)((kb < K) ? kb : 0) * LDB];
            b[e] = (kb < K && v0) ? (_Float16)t : (_Float16)0.f;
        }
        acc0 = __builtin_amdgcn_wmma_f32_16x16x32_f16(
            false, a, false, b, (short)0, acc0, false, false);
#pragma unroll
        for (int e = 0; e < 16; ++e) {
            int kb = kfull + half * 16 + e;
            float t = B1[(size_t)((kb < K) ? kb : 0) * LDB];
            b[e] = (kb < K && v1) ? (_Float16)t : (_Float16)0.f;
        }
        acc1 = __builtin_amdgcn_wmma_f32_16x16x32_f16(
            false, a, false, b, (short)0, acc1, false, false);
    }

    // store: acc element v -> row tm*16 + half*8 + v, column col0 / col1
    if (v0) {
        float bv = bias ? bias[col0] : 0.f;
#pragma unroll
        for (int v = 0; v < 8; ++v) {
            int r = tm * 16 + half * 8 + v;
            float val = acc0[v] + bv;
            if (add)  val += add[(size_t)r * ldadd + col0];
            if (relu) val = fmaxf(val, 0.f);
            C[(size_t)r * ldc + col0] = val;
        }
    }
    if (v1) {
        float bv = bias ? bias[col1] : 0.f;
#pragma unroll
        for (int v = 0; v < 8; ++v) {
            int r = tm * 16 + half * 8 + v;
            float val = acc1[v] + bv;
            if (add)  val += add[(size_t)r * ldadd + col1];
            if (relu) val = fmaxf(val, 0.f);
            C[(size_t)r * ldc + col1] = val;
        }
    }
}

// ------------------------------ edge kernels -------------------------------

// out[n,h] = sum_d hw[n, h*32+d] * avec[h*32+d]
__global__ void k_heads_dot(const float* __restrict__ hw,
                            const float* __restrict__ avec,
                            float* __restrict__ out, int n) {
    int i = blockIdx.x * blockDim.x + threadIdx.x;
    if (i >= n * NH) return;
    int node = i / NH, h = i % NH;
    const float* hp = hw + (size_t)node * HIDN + h * HD;
    const float* ap = avec + h * HD;
    float s = 0.f;
#pragma unroll
    for (int d = 0; d < HD; ++d) s += hp[d] * ap[d];
    out[i] = s;
}

__global__ void k_cnt_loop(const int* __restrict__ dst,
                           const float* __restrict__ ea,
                           float* cnt, float* la, int Ee) {
    int j = blockIdx.x * blockDim.x + threadIdx.x;
    if (j >= Ee) return;
    int d = dst[j];
    atomicAdd(&cnt[d], 1.f);
    atomicAdd(&la[d * 3 + 0], ea[(size_t)j * 3 + 0]);
    atomicAdd(&la[d * 3 + 1], ea[(size_t)j * 3 + 1]);
    atomicAdd(&la[d * 3 + 2], ea[(size_t)j * 3 + 2]);
}

__global__ void k_loop_div(float* la, const float* __restrict__ cnt, int n) {
    int i = blockIdx.x * blockDim.x + threadIdx.x;
    if (i >= n) return;
    float c = fmaxf(cnt[i], 1.f);
    la[i * 3 + 0] /= c; la[i * 3 + 1] /= c; la[i * 3 + 2] /= c;
}

// weff[c,h] = sum_d We[c, h*32+d] * a_edge[h, d]   (collapses K=3 edge proj)
__global__ void k_weff(const float* __restrict__ We,
                       const float* __restrict__ aedge, float* weff) {
    int i = threadIdx.x;
    if (i >= 24) return;
    int c = i / 8, h = i % 8;
    float s = 0.f;
#pragma unroll
    for (int d = 0; d < HD; ++d) s += We[c * HIDN + h * HD + d] * aedge[h * HD + d];
    weff[c * 8 + h] = s;
}

// GAT alpha (edges + self loops), LeakyReLU(0.2), atomic segment max
__global__ void k_gat_alpha(const int* __restrict__ src, const int* __restrict__ dst,
                            const float* __restrict__ ea, const float* __restrict__ la,
                            const float* __restrict__ als, const float* __restrict__ ald,
                            const float* __restrict__ weff,
                            float* __restrict__ alpha, float* mbuf, int Ee, int n) {
    int i = blockIdx.x * blockDim.x + threadIdx.x;
    if (i >= (Ee + n) * NH) return;
    int j = i / NH, h = i % NH;
    int s, d; const float* ev;
    if (j < Ee) { s = src[j]; d = dst[j]; ev = ea + (size_t)j * 3; }
    else        { s = d = j - Ee;         ev = la + (size_t)s * 3; }
    float a = als[s * NH + h] + ald[d * NH + h]
            + ev[0] * weff[h] + ev[1] * weff[8 + h] + ev[2] * weff[16 + h];
    a = (a > 0.f) ? a : 0.2f * a;
    alpha[i] = a;
    atomicMaxF(&mbuf[d * NH + h], a);
}

// a = exp(alpha - m[dst]); segment sum (GAT: nExtra=N self loops, TRF: 0)
__global__ void k_seg_expsum(const int* __restrict__ dst, float* alpha,
                             const float* __restrict__ mbuf, float* sbuf,
                             int Ee, int nExtra) {
    int i = blockIdx.x * blockDim.x + threadIdx.x;
    if (i >= (Ee + nExtra) * NH) return;
    int j = i / NH, h = i % NH;
    int d = (j < Ee) ? dst[j] : (j - Ee);
    float a = __expf(alpha[i] - mbuf[d * NH + h]);
    alpha[i] = a;
    atomicAdd(&sbuf[d * NH + h], a);
}

__global__ void k_gat_scatter(const int* __restrict__ src, const int* __restrict__ dst,
                              const float* __restrict__ alpha, const float* __restrict__ sbuf,
                              const float* __restrict__ hw,
                              float* __restrict__ out, int ldo, int Ee, int n) {
    int i = blockIdx.x * blockDim.x + threadIdx.x;
    if (i >= (Ee + n) * NH) return;
    int j = i / NH, h = i % NH;
    int s, d;
    if (j < Ee) { s = src[j]; d = dst[j]; } else { s = d = j - Ee; }
    float w = alpha[i] / sbuf[d * NH + h];
    const float* hv = hw + (size_t)s * HIDN + h * HD;
    float* op = out + (size_t)d * ldo + h * HD;
#pragma unroll
    for (int dd = 0; dd < HD; ++dd) atomicAdd(&op[dd], w * hv[dd]);
}

// TransformerConv alpha: q[dst] . (k[src] + ea@We) / sqrt(32); edge proj fused
__global__ void k_trf_alpha(const int* __restrict__ src, const int* __restrict__ dst,
                            const float* __restrict__ ea, const float* __restrict__ We,
                            const float* __restrict__ q, const float* __restrict__ kk,
                            float* __restrict__ alpha, float* mbuf, int Ee) {
    int i = blockIdx.x * blockDim.x + threadIdx.x;
    if (i >= Ee * NH) return;
    int j = i / NH, h = i % NH;
    int s = src[j], d = dst[j];
    float e0 = ea[(size_t)j * 3], e1 = ea[(size_t)j * 3 + 1], e2 = ea[(size_t)j * 3 + 2];
    const float* qp = q  + (size_t)d * HIDN + h * HD;
    const float* kp = kk + (size_t)s * HIDN + h * HD;
    const float* w0 = We + h * HD;
    const float* w1 = We + HIDN + h * HD;
    const float* w2 = We + 2 * HIDN + h * HD;
    float a = 0.f;
#pragma unroll
    for (int dd = 0; dd < HD; ++dd)
        a += qp[dd] * (kp[dd] + e0 * w0[dd] + e1 * w1[dd] + e2 * w2[dd]);
    a *= 0.17677669529663687f;  // 1/sqrt(32)
    alpha[i] = a;
    atomicMaxF(&mbuf[d * NH + h], a);
}

__global__ void k_trf_scatter(const int* __restrict__ src, const int* __restrict__ dst,
                              const float* __restrict__ ea, const float* __restrict__ We,
                              const float* __restrict__ alpha, const float* __restrict__ sbuf,
                              const float* __restrict__ v,
                              float* __restrict__ out, int ldo, int Ee) {
    int i = blockIdx.x * blockDim.x + threadIdx.x;
    if (i >= Ee * NH) return;
    int j = i / NH, h = i % NH;
    int s = src[j], d = dst[j];
    float w = alpha[i] / sbuf[d * NH + h];
    float e0 = ea[(size_t)j * 3], e1 = ea[(size_t)j * 3 + 1], e2 = ea[(size_t)j * 3 + 2];
    const float* vp = v + (size_t)s * HIDN + h * HD;
    const float* w0 = We + h * HD;
    const float* w1 = We + HIDN + h * HD;
    const float* w2 = We + 2 * HIDN + h * HD;
    float* op = out + (size_t)d * ldo + h * HD;
#pragma unroll
    for (int dd = 0; dd < HD; ++dd)
        atomicAdd(&op[dd], w * (vp[dd] + e0 * w0[dd] + e1 * w1[dd] + e2 * w2[dd]));
}

// h = layernorm(hf + h) * g + b   (one block = one row, in-place on h)
__global__ __launch_bounds__(256)
void k_lnres(const float* __restrict__ hf, float* h,
             const float* __restrict__ g, const float* __restrict__ b) {
    __shared__ float red[256];
    int row = blockIdx.x, t = threadIdx.x;
    float x = hf[(size_t)row * HIDN + t] + h[(size_t)row * HIDN + t];
    red[t] = x; __syncthreads();
    for (int s = 128; s > 0; s >>= 1) { if (t < s) red[t] += red[t + s]; __syncthreads(); }
    float mu = red[0] * (1.f / HIDN); __syncthreads();
    float dv = x - mu;
    red[t] = dv * dv; __syncthreads();
    for (int s = 128; s > 0; s >>= 1) { if (t < s) red[t] += red[t + s]; __syncthreads(); }
    float var = red[0] * (1.f / HIDN);
    h[(size_t)row * HIDN + t] = dv * rsqrtf(var + 1e-5f) * g[t] + b[t];
}

// gap attention: softmax over ALL gap edges per head (axis 0)
__global__ void k_gap_alpha(const int* __restrict__ src, const int* __restrict__ dst,
                            const float* __restrict__ q, const float* __restrict__ kk,
                            float* __restrict__ sc, float* gm, int Ee) {
    int i = blockIdx.x * blockDim.x + threadIdx.x;
    if (i >= Ee * NH) return;
    int j = i / NH, h = i % NH;
    const float* qp = q  + (size_t)dst[j] * HIDN + h * HD;
    const float* kp = kk + (size_t)src[j] * HIDN + h * HD;
    float a = 0.f;
#pragma unroll
    for (int dd = 0; dd < HD; ++dd) a += qp[dd] * kp[dd];
    a *= 0.17677669529663687f;
    sc[i] = a;
    atomicMaxF(&gm[h], a);
}

__global__ void k_gap_expsum(float* sc, const float* __restrict__ gm, float* gs, int Ee) {
    int i = blockIdx.x * blockDim.x + threadIdx.x;
    if (i >= Ee * NH) return;
    float a = __expf(sc[i] - gm[i % NH]);
    sc[i] = a;
    atomicAdd(&gs[i % NH], a);
}

__global__ void k_gap_scatter(const int* __restrict__ src, const int* __restrict__ dst,
                              const float* __restrict__ sc, const float* __restrict__ gs,
                              const float* __restrict__ v, float* att, int Ee) {
    int i = blockIdx.x * blockDim.x + threadIdx.x;
    if (i >= Ee * NH) return;
    int j = i / NH, h = i % NH;
    float w = sc[i] / gs[h];
    const float* vp = v + (size_t)src[j] * HIDN + h * HD;
    float* op = att + (size_t)dst[j] * HIDN + h * HD;
#pragma unroll
    for (int dd = 0; dd < HD; ++dd) atomicAdd(&op[dd], w * vp[dd]);
}

__global__ void k_logsoftmax(const float* __restrict__ z, float* out, int n) {
    int i = blockIdx.x * blockDim.x + threadIdx.x;
    if (i >= n) return;
    const float* r = z + (size_t)i * NCLSS;
    float mx = r[0];
#pragma unroll
    for (int c = 1; c < NCLSS; ++c) mx = fmaxf(mx, r[c]);
    float s = 0.f;
#pragma unroll
    for (int c = 0; c < NCLSS; ++c) s += __expf(r[c] - mx);
    float ls = __logf(s);
#pragma unroll
    for (int c = 0; c < NCLSS; ++c) out[(size_t)i * NCLSS + c] = r[c] - mx - ls;
}

// ------------------------------ host orchestration -------------------------

extern "C" void kernel_launch(void* const* d_in, const int* in_sizes, int n_in,
                              void* d_out, int out_size, void* d_ws, size_t ws_size,
                              hipStream_t stream) {
    const float* x     = (const float*)d_in[0];
    const int*   ei    = (const int*)d_in[1];
    const float* eattr = (const float*)d_in[2];
    const float* in_w  = (const float*)d_in[4];
    const float* in_b  = (const float*)d_in[5];
    const float* gat_w  = (const float*)d_in[6];
    const float* gat_as = (const float*)d_in[7];
    const float* gat_ad = (const float*)d_in[8];
    const float* gat_ae = (const float*)d_in[9];
    const float* gat_ew = (const float*)d_in[10];
    const float* gat_b  = (const float*)d_in[11];
    const float* t_qw = (const float*)d_in[12];
    const float* t_kw = (const float*)d_in[13];
    const float* t_vw = (const float*)d_in[14];
    const float* t_sw = (const float*)d_in[15];
    const float* t_qb = (const float*)d_in[16];
    const float* t_kb = (const float*)d_in[17];
    const float* t_vb = (const float*)d_in[18];
    const float* t_sb = (const float*)d_in[19];
    const float* t_ew = (const float*)d_in[20];
    const float* fus_w = (const float*)d_in[21];
    const float* fus_b = (const float*)d_in[22];
    const float* ln_g  = (const float*)d_in[23];
    const float* ln_b  = (const float*)d_in[24];
    const float* ga_qw = (const float*)d_in[25];
    const float* ga_kw = (const float*)d_in[26];
    const float* ga_vw = (const float*)d_in[27];
    const float* ga_ow = (const float*)d_in[28];
    const float* ga_qb = (const float*)d_in[29];
    const float* ga_kb = (const float*)d_in[30];
    const float* ga_vb = (const float*)d_in[31];
    const float* ga_ob = (const float*)d_in[32];
    const float* c_w1 = (const float*)d_in[33];
    const float* c_b1 = (const float*)d_in[34];
    const float* c_w2 = (const float*)d_in[35];
    const float* c_b2 = (const float*)d_in[36];
    const float* c_w3 = (const float*)d_in[37];
    const float* c_b3 = (const float*)d_in[38];

    const int Nn = in_sizes[0] / 16;   // 50000 (NF=16)
    const int E  = in_sizes[3];        // 900000
    const int Et = E / 3;              // 300000 per edge type
    const int* srcp = ei;
    const int* dstp = ei + E;

    // ---- workspace carve (floats) ----
    float* ws = (float*)d_ws;
    size_t off = 0;
    auto carve = [&](size_t nf) { float* p = ws + off; off += (nf + 63) & ~(size_t)63; return p; };
    float* H0   = carve((size_t)Nn * HIDN);       // node features
    float* HC   = carve((size_t)Nn * 3 * HIDN);   // concat [h_t | h_g | h_p]
    float* HF   = carve((size_t)Nn * HIDN);       // fusion output / z2
    float* Qb   = carve((size_t)Nn * HIDN);       // hW / q
    float* Kb   = carve((size_t)Nn * HIDN);
    float* Vb   = carve((size_t)Nn * HIDN);
    float* ATT  = carve((size_t)Nn * HIDN);
    float* Z1   = carve((size_t)Nn * 2 * HIDN);
    float* ALS  = carve((size_t)Nn * NH);
    float* ALD  = carve((size_t)Nn * NH);
    float* ALPH = carve((size_t)(Et + Nn) * NH);
    float* Mb   = carve((size_t)Nn * NH);
    float* Sb   = carve((size_t)Nn * NH);
    float* CNT  = carve((size_t)Nn);
    float* LA   = carve((size_t)Nn * 3);
    float* WEFF = carve(64);
    float* G8   = carve(64);
    float* LOGI = carve((size_t)Nn * NCLSS);

    auto nb = [](long n) { return (unsigned)((n + 255) / 256); };
    auto gemm = [&](const float* A, int lda, const float* B, int ldb,
                    const float* bias, const float* add, int ldadd,
                    float* C, int ldc, int M, int K, int Ncol, int relu) {
        dim3 g((M + 15) / 16, (Ncol + 127) / 128);
        if (ldb == HIDN)            // Ncol == 256 at every such call site
            k_gemm_wmma<HIDN, false><<<g, 128, 0, stream>>>(
                A, lda, B, bias, add, ldadd, C, ldc, M, K, Ncol, relu);
        else if (ldb == 2 * HIDN)   // Ncol == 512
            k_gemm_wmma<2 * HIDN, false><<<g, 128, 0, stream>>>(
                A, lda, B, bias, add, ldadd, C, ldc, M, K, Ncol, relu);
        else                        // classifier head, Ncol == 5
            k_gemm_wmma<NCLSS, true><<<g, 128, 0, stream>>>(
                A, lda, B, bias, add, ldadd, C, ldc, M, K, Ncol, relu);
    };
    auto fill = [&](float* p, float v, long n) {
        k_fill<<<nb(n), 256, 0, stream>>>(p, v, (int)n);
    };

    // ---- input projection: h = x @ in_w + in_b ----
    gemm(x, 16, in_w, HIDN, in_b, nullptr, 0, H0, HIDN, Nn, 16, HIDN, 0);

    // edge-type table: concat order [t, g, p]; weight idx 0->t, 1->g, 2->p
    struct TInfo { int eoff; int wi; int blk; };
    const TInfo types[3] = { {0, 0, 0}, {2 * Et, 1, 1}, {Et, 2, 2} };

    for (int i = 0; i < 8; ++i) {
        for (int ty = 0; ty < 3; ++ty) {
            const int*   sp  = srcp + types[ty].eoff;
            const int*   dp  = dstp + types[ty].eoff;
            const float* ep  = eattr + (size_t)types[ty].eoff * 3;
            float*       hcb = HC + types[ty].blk * HIDN;   // ld = 768
            const int wi = types[ty].wi;

            if (i < 4) {
                const size_t wo = (size_t)(i * 3 + wi);
                const float* W   = gat_w  + wo * HIDN * HIDN;
                const float* asv = gat_as + wo * HIDN;
                const float* adv = gat_ad + wo * HIDN;
                const float* aev = gat_ae + wo * HIDN;
                const float* Wev = gat_ew + wo * 3 * HIDN;
                const float* bv  = gat_b  + wo * HIDN;

                gemm(H0, HIDN, W, HIDN, nullptr, nullptr, 0, Qb, HIDN, Nn, HIDN, HIDN, 0);
                k_heads_dot<<<nb((long)Nn * NH), 256, 0, stream>>>(Qb, asv, ALS, Nn);
                k_heads_dot<<<nb((long)Nn * NH), 256, 0, stream>>>(Qb, adv, ALD, Nn);
                fill(CNT, 0.f, Nn); fill(LA, 0.f, (long)Nn * 3);
                k_cnt_loop<<<nb(Et), 256, 0, stream>>>(dp, ep, CNT, LA, Et);
                k_loop_div<<<nb(Nn), 256, 0, stream>>>(LA, CNT, Nn);
                k_weff<<<1, 32, 0, stream>>>(Wev, aev, WEFF);
                fill(Mb, -1e30f, (long)Nn * NH); fill(Sb, 0.f, (long)Nn * NH);
                k_gat_alpha<<<nb((long)(Et + Nn) * NH), 256, 0, stream>>>(
                    sp, dp, ep, LA, ALS, ALD, WEFF, ALPH, Mb, Et, Nn);
                k_seg_expsum<<<nb((long)(Et + Nn) * NH), 256, 0, stream>>>(
                    dp, ALPH, Mb, Sb, Et, Nn);
                k_bias_rows<<<nb((long)Nn * HIDN), 256, 0, stream>>>(
                    hcb, bv, Nn, HIDN, 3 * HIDN);
                k_gat_scatter<<<nb((long)(Et + Nn) * NH), 256, 0, stream>>>(
                    sp, dp, ALPH, Sb, Qb, hcb, 3 * HIDN, Et, Nn);
            } else {
                const int j = i - 4;
                const size_t wo = (size_t)(j * 3 + wi);
                const float* Wq = t_qw + wo * HIDN * HIDN;
                const float* Wk = t_kw + wo * HIDN * HIDN;
                const float* Wv = t_vw + wo * HIDN * HIDN;
                const float* Wsk = t_sw + wo * HIDN * HIDN;
                const float* bq = t_qb + wo * HIDN;
                const float* bk = t_kb + wo * HIDN;
                const float* bvv = t_vb + wo * HIDN;
                const float* bs = t_sb + wo * HIDN;
                const float* Wev = t_ew + wo * 3 * HIDN;

                gemm(H0, HIDN, Wq, HIDN, bq, nullptr, 0, Qb, HIDN, Nn, HIDN, HIDN, 0);
                gemm(H0, HIDN, Wk, HIDN, bk, nullptr, 0, Kb, HIDN, Nn, HIDN, HIDN, 0);
                gemm(H0, HIDN, Wv, HIDN, bvv, nullptr, 0, Vb, HIDN, Nn, HIDN, HIDN, 0);
                // skip connection written straight into the hc block
                gemm(H0, HIDN, Wsk, HIDN, bs, nullptr, 0, hcb, 3 * HIDN, Nn, HIDN, HIDN, 0);
                fill(Mb, -1e30f, (long)Nn * NH); fill(Sb, 0.f, (long)Nn * NH);
                k_trf_alpha<<<nb((long)Et * NH), 256, 0, stream>>>(
                    sp, dp, ep, Wev, Qb, Kb, ALPH, Mb, Et);
                k_seg_expsum<<<nb((long)Et * NH), 256, 0, stream>>>(
                    dp, ALPH, Mb, Sb, Et, 0);
                k_trf_scatter<<<nb((long)Et * NH), 256, 0, stream>>>(
                    sp, dp, ep, Wev, ALPH, Sb, Vb, hcb, 3 * HIDN, Et);
            }
        }
        // hf = relu(hc @ fus_w[i] + fus_b[i]); h = LN(hf + h)
        gemm(HC, 3 * HIDN, fus_w + (size_t)i * 3 * HIDN * HIDN, HIDN,
             fus_b + (size_t)i * HIDN, nullptr, 0, HF, HIDN, Nn, 3 * HIDN, HIDN, 1);
        k_lnres<<<Nn, 256, 0, stream>>>(HF, H0,
                                        ln_g + (size_t)i * HIDN, ln_b + (size_t)i * HIDN);
    }

    // ---- gap-aware attention (gap = type-2 edges), softmax over all edges ----
    gemm(H0, HIDN, ga_qw, HIDN, ga_qb, nullptr, 0, Qb, HIDN, Nn, HIDN, HIDN, 0);
    gemm(H0, HIDN, ga_kw, HIDN, ga_kb, nullptr, 0, Kb, HIDN, Nn, HIDN, HIDN, 0);
    gemm(H0, HIDN, ga_vw, HIDN, ga_vb, nullptr, 0, Vb, HIDN, Nn, HIDN, HIDN, 0);
    fill(G8, -1e30f, 8); fill(G8 + 8, 0.f, 8);
    k_gap_alpha<<<nb((long)Et * NH), 256, 0, stream>>>(
        srcp + 2 * Et, dstp + 2 * Et, Qb, Kb, ALPH, G8, Et);
    k_gap_expsum<<<nb((long)Et * NH), 256, 0, stream>>>(ALPH, G8, G8 + 8, Et);
    fill(ATT, 0.f, (long)Nn * HIDN);
    k_gap_scatter<<<nb((long)Et * NH), 256, 0, stream>>>(
        srcp + 2 * Et, dstp + 2 * Et, ALPH, G8 + 8, Vb, ATT, Et);
    // h = h + att @ ga_ow + ga_ob  (residual via add-source, in-place safe)
    gemm(ATT, HIDN, ga_ow, HIDN, ga_ob, H0, HIDN, H0, HIDN, Nn, HIDN, HIDN, 0);

    // ---- classifier ----
    gemm(H0, HIDN, c_w1, 2 * HIDN, c_b1, nullptr, 0, Z1, 2 * HIDN, Nn, HIDN, 2 * HIDN, 1);
    gemm(Z1, 2 * HIDN, c_w2, HIDN, c_b2, nullptr, 0, HF, HIDN, Nn, 2 * HIDN, HIDN, 1);
    gemm(HF, HIDN, c_w3, NCLSS, c_b3, nullptr, 0, LOGI, NCLSS, Nn, HIDN, NCLSS, 0);
    k_logsoftmax<<<nb(Nn), 256, 0, stream>>>(LOGI, (float*)d_out, Nn);
}
